// Decoder_32710470926956
// MI455X (gfx1250) — matrix operand order repaired and checked by reading, not compiled, not run on hardware
//
#include <hip/hip_runtime.h>
#include <hip/hip_bf16.h>

// ---------------------------------------------------------------------------
// Types
// ---------------------------------------------------------------------------
typedef __bf16 bf16;
typedef __attribute__((ext_vector_type(16))) __bf16 v16bf;
typedef __attribute__((ext_vector_type(8)))  __bf16 v8bf;
typedef __attribute__((ext_vector_type(4)))  __bf16 v4bf;
typedef __attribute__((ext_vector_type(2)))  __bf16 v2bf;
typedef __attribute__((ext_vector_type(8)))  float  v8f;

#define LDSPAD 40   // LDS row stride in bf16 elements (80B, 16B-aligned, bank-skewed)

__device__ __forceinline__ bf16 f2bf(float f) {
    unsigned int u = __builtin_bit_cast(unsigned int, f);
    unsigned int r = (u + 0x7FFFu + ((u >> 16) & 1u)) >> 16;   // RNE
    return __builtin_bit_cast(bf16, (unsigned short)r);
}

// contiguous 16 bf16 (two 16B chunks)
__device__ __forceinline__ v16bf load_frag16(const bf16* p) {
    v8bf lo = *reinterpret_cast<const v8bf*>(p);
    v8bf hi = *reinterpret_cast<const v8bf*>(p + 8);
    v16bf r;
#pragma unroll
    for (int i = 0; i < 8; ++i) { r[i] = lo[i]; r[i + 8] = hi[i]; }
    return r;
}

// A-matrix fragment (16x32 bf16), contiguous k order (B frags use the same):
//   lane: m = lane&15, half = lane>>4
//   e=0..15 -> k = half*16 + e
__device__ __forceinline__ v16bf load_a_frag(const bf16* row, int half) {
    v8bf lo = *reinterpret_cast<const v8bf*>(row + half * 16);
    v8bf hi = *reinterpret_cast<const v8bf*>(row + half * 16 + 8);
    v16bf r;
#pragma unroll
    for (int i = 0; i < 8; ++i) { r[i] = lo[i]; r[i + 8] = hi[i]; }
    return r;
}

__device__ __forceinline__ v8f wmma_bf16(v16bf a, v16bf b, v8f c) {
    return __builtin_amdgcn_wmma_f32_16x16x32_bf16(false, a, false, b,
                                                   (short)0, c, false, false);
}

// ---------------------------------------------------------------------------
// Conversion / packing kernels
// ---------------------------------------------------------------------------
__global__ void f32_to_bf16_kernel(const float* __restrict__ in,
                                   bf16* __restrict__ out, size_t n) {
    size_t i = (size_t)blockIdx.x * blockDim.x + threadIdx.x;
    if (i < n) out[i] = f2bf(in[i]);
}

// out[k][j] for j in [0, n_srcs*n_each): columns from a|b|c side by side
__global__ void pack_bf16_kernel(const float* __restrict__ a,
                                 const float* __restrict__ b,
                                 const float* __restrict__ c,
                                 bf16* __restrict__ out,
                                 int K, int n_each, int n_srcs) {
    int i = blockIdx.x * blockDim.x + threadIdx.x;
    int N = n_each * n_srcs;
    if (i >= K * N) return;
    int k = i / N, j = i % N;
    const float* src = (j < n_each) ? a : (j < 2 * n_each ? b : c);
    out[i] = f2bf(src[k * n_each + (j % n_each)]);
}

// W_sum[p][j] = sum_h w_o[64h+p][j]   (tile(head) @ w_o == head @ W_sum)
__global__ void wsum_kernel(const float* __restrict__ w_o, bf16* __restrict__ out) {
    int i = blockIdx.x * blockDim.x + threadIdx.x;
    if (i >= 64 * 512) return;
    int p = i / 512, j = i % 512;
    float s = 0.f;
#pragma unroll
    for (int h = 0; h < 8; ++h) s += w_o[(size_t)(h * 64 + p) * 512 + j];
    out[i] = f2bf(s);
}

__global__ void copy_f32_kernel(const float* __restrict__ in,
                                float* __restrict__ out, size_t n) {
    size_t i = (size_t)blockIdx.x * blockDim.x + threadIdx.x;
    if (i < n) out[i] = in[i];
}

// ---------------------------------------------------------------------------
// Generic bf16 WMMA GEMM:  out[M,N] = X[M,K] @ W[K,N]  (X row-major ld=K)
// Block = 256 threads = 8 waves; each wave owns 16 rows, block covers 128x64.
// W tile (32 x 64) staged transposed in double-buffered LDS (packed b32
// stores); one barrier per K-step, staging overlaps WMMA of previous tile.
// EPI: 0 = f32 store, 1 = bf16 store, 2 = +bias,LeakyReLU,bf16, 3 = +bias,f32
// ---------------------------------------------------------------------------
template<int EPI>
__global__ __launch_bounds__(256)
void gemm_bf16_kernel(const bf16* __restrict__ X, const bf16* __restrict__ W,
                      const float* __restrict__ bias,
                      float* __restrict__ outF, bf16* __restrict__ outB,
                      int M, int N, int K) {
    __shared__ bf16 Wt[2][64][LDSPAD];
    const int tid = threadIdx.x;
    const int lane = tid & 31, wid = tid >> 5;
    const int half = lane >> 4, ln16 = lane & 15;
    const int m0 = blockIdx.x * 128 + wid * 16;
    const int nb = blockIdx.y * 64;
    const bf16* xrow = X + (size_t)(m0 + ln16) * K;
    const int sn0 = (tid & 15) * 4;     // staging: 2(k) x 4(n) sub-tile
    const int sk0 = (tid >> 4) * 2;

    auto stageW = [&](int buf, int kb) {
        const bf16* g = W + (size_t)(kb + sk0) * N + nb + sn0;
        v4bf lo = *reinterpret_cast<const v4bf*>(g);
        v4bf hi = *reinterpret_cast<const v4bf*>(g + N);
#pragma unroll
        for (int j = 0; j < 4; ++j) {
            v2bf p; p[0] = lo[j]; p[1] = hi[j];
            *reinterpret_cast<v2bf*>(&Wt[buf][sn0 + j][sk0]) = p;  // b32 store
        }
    };

    v8f acc0{}, acc1{}, acc2{}, acc3{};
    int buf = 0;
    stageW(0, 0);
    for (int kb = 0; kb < K; kb += 32) {
        __syncthreads();
        if (kb + 32 < K) {
            stageW(buf ^ 1, kb + 32);
            if (kb + 64 < K)
                __builtin_prefetch(W + (size_t)(kb + 64 + sk0) * N + nb + sn0, 0, 1);
        }
        v16bf a = load_a_frag(xrow + kb, half);
        acc0 = wmma_bf16(a, load_frag16(&Wt[buf][ 0 + ln16][half * 16]), acc0);
        acc1 = wmma_bf16(a, load_frag16(&Wt[buf][16 + ln16][half * 16]), acc1);
        acc2 = wmma_bf16(a, load_frag16(&Wt[buf][32 + ln16][half * 16]), acc2);
        acc3 = wmma_bf16(a, load_frag16(&Wt[buf][48 + ln16][half * 16]), acc3);
        buf ^= 1;
    }
#pragma unroll
    for (int r = 0; r < 8; ++r) {
        int row = m0 + r + 8 * half;
        float v[4] = {acc0[r], acc1[r], acc2[r], acc3[r]};
#pragma unroll
        for (int j = 0; j < 4; ++j) {
            int col = nb + j * 16 + ln16;
            float val = v[j];
            if (EPI >= 2) val += bias[col];
            if (EPI == 2) val = val >= 0.f ? val : 0.01f * val;
            size_t o = (size_t)row * N + col;
            if (EPI == 0 || EPI == 3) outF[o] = val;
            else                      outB[o] = f2bf(val);
        }
    }
}

// ---------------------------------------------------------------------------
// Flash attention (online softmax), head dim 64, bf16 in / bf16 out.
// Block = 8 waves x 16 query rows = 128 rows. Key loop steps 32 keys;
// V tile staged transposed in double-buffered LDS by the whole block (packed
// b32 stores, one barrier per step); per-wave P-tile goes through a private
// LDS scratch to convert C-layout f32 -> A-layout bf16.
// ---------------------------------------------------------------------------
template<bool MASKED>
__global__ __launch_bounds__(256)
void attn_kernel(const bf16* __restrict__ Q, int ldq,
                 const bf16* __restrict__ Kp, const bf16* __restrict__ V, int ldkv,
                 bf16* __restrict__ Out, int T) {
    __shared__ bf16 Vt[2][64][LDSPAD];     // Vt[buf][col][key-in-tile]
    __shared__ bf16 Psh[8][16][LDSPAD];    // per-wave P scratch [row][key]
    const int tid = threadIdx.x;
    const int lane = tid & 31, wid = tid >> 5;
    const int half = lane >> 4, ln16 = lane & 15;
    const int q0 = blockIdx.x * 128 + wid * 16;
    const int sc0 = (tid & 15) * 4;        // staging: 2(key) x 4(col) sub-tile
    const int sr0 = (tid >> 4) * 2;

    auto stageV = [&](int buf, int kb) {
        const bf16* g = V + (size_t)(kb + sr0) * ldkv + sc0;
        v4bf lo = *reinterpret_cast<const v4bf*>(g);
        v4bf hi = *reinterpret_cast<const v4bf*>(g + ldkv);
#pragma unroll
        for (int j = 0; j < 4; ++j) {
            v2bf p; p[0] = lo[j]; p[1] = hi[j];
            *reinterpret_cast<v2bf*>(&Vt[buf][sc0 + j][sr0]) = p;  // b32 store
        }
    };

    const bf16* qrow = Q + (size_t)(q0 + ln16) * ldq;
    v16bf qa0 = load_a_frag(qrow, half);        // k = 0..31
    v16bf qa1 = load_a_frag(qrow + 32, half);   // k = 32..63

    float mrow[8], lrow[8];
    v8f acc0{}, acc1{}, acc2{}, acc3{};
#pragma unroll
    for (int r = 0; r < 8; ++r) { mrow[r] = -__builtin_inff(); lrow[r] = 0.f; }

    const int tend = MASKED ? (blockIdx.x * 128 + 128) : T;
    const float scale = 0.125f;   // 1/sqrt(64)

    int buf = 0;
    stageV(0, 0);
    for (int kb = 0; kb < tend; kb += 32) {
        __syncthreads();
        if (kb + 32 < tend) {
            stageV(buf ^ 1, kb + 32);
            __builtin_prefetch(Kp + (size_t)(kb + 32 + ln16) * ldkv, 0, 1);
        }
        if (!(MASKED && kb > q0 + 15)) {
            // S = (Q K^T) * scale for keys kb..kb+31 (two 16-col tiles)
            const bf16* k0 = Kp + (size_t)(kb + ln16) * ldkv;
            const bf16* k1 = Kp + (size_t)(kb + 16 + ln16) * ldkv;
            v8f s0{}, s1{};
            s0 = wmma_bf16(qa0, load_frag16(k0 + half * 16), s0);
            s0 = wmma_bf16(qa1, load_frag16(k0 + 32 + half * 16), s0);
            s1 = wmma_bf16(qa0, load_frag16(k1 + half * 16), s1);
            s1 = wmma_bf16(qa1, load_frag16(k1 + 32 + half * 16), s1);

            float p0s[8], p1s[8];
#pragma unroll
            for (int r = 0; r < 8; ++r) {
                float a0 = s0[r] * scale, a1 = s1[r] * scale;
                if (MASKED) {
                    int qr = q0 + r + 8 * half;
                    if (kb + ln16 > qr)      a0 = -__builtin_inff();
                    if (kb + 16 + ln16 > qr) a1 = -__builtin_inff();
                }
                float mx = fmaxf(a0, a1);
#pragma unroll
                for (int o = 8; o >= 1; o >>= 1) mx = fmaxf(mx, __shfl_xor(mx, o, 32));
                float mnew  = fmaxf(mrow[r], mx);
                float alpha = __expf(mrow[r] - mnew);
                float p0 = __expf(a0 - mnew);
                float p1 = __expf(a1 - mnew);
                float ps = p0 + p1;
#pragma unroll
                for (int o = 8; o >= 1; o >>= 1) ps += __shfl_xor(ps, o, 32);
                lrow[r] = lrow[r] * alpha + ps;
                mrow[r] = mnew;
                acc0[r] *= alpha; acc1[r] *= alpha; acc2[r] *= alpha; acc3[r] *= alpha;
                p0s[r] = p0; p1s[r] = p1;
            }

            // C-layout -> A-layout via wave-private LDS (DS ops in-order per wave)
#pragma unroll
            for (int r = 0; r < 8; ++r) {
                Psh[wid][r + 8 * half][ln16]      = f2bf(p0s[r]);
                Psh[wid][r + 8 * half][16 + ln16] = f2bf(p1s[r]);
            }
            v16bf pa = load_a_frag(&Psh[wid][ln16][0], half);
            acc0 = wmma_bf16(pa, load_frag16(&Vt[buf][ 0 + ln16][half * 16]), acc0);
            acc1 = wmma_bf16(pa, load_frag16(&Vt[buf][16 + ln16][half * 16]), acc1);
            acc2 = wmma_bf16(pa, load_frag16(&Vt[buf][32 + ln16][half * 16]), acc2);
            acc3 = wmma_bf16(pa, load_frag16(&Vt[buf][48 + ln16][half * 16]), acc3);
        }
        buf ^= 1;
    }

#pragma unroll
    for (int r = 0; r < 8; ++r) {
        float inv = 1.0f / lrow[r];
        bf16* o = Out + (size_t)(q0 + r + 8 * half) * 64;
        o[ 0 + ln16] = f2bf(acc0[r] * inv);
        o[16 + ln16] = f2bf(acc1[r] * inv);
        o[32 + ln16] = f2bf(acc2[r] * inv);
        o[48 + ln16] = f2bf(acc3[r] * inv);
    }
}

// ---------------------------------------------------------------------------
// Residual add + LayerNorm over D=512. One block (256 threads) per row.
// DUAL also emits a bf16 copy (feeds the next GEMM).
// ---------------------------------------------------------------------------
template<bool DUAL>
__global__ __launch_bounds__(256)
void ln_residual_kernel(const float* __restrict__ a, const float* __restrict__ b,
                        const float* __restrict__ g, const float* __restrict__ beta,
                        float* __restrict__ outF, bf16* __restrict__ outB) {
    __shared__ float red[16];
    const int row = blockIdx.x, tid = threadIdx.x;
    const size_t base = (size_t)row * 512;
    float x0 = a[base + tid] + b[base + tid];
    float x1 = a[base + tid + 256] + b[base + tid + 256];
    float s = x0 + x1, s2 = x0 * x0 + x1 * x1;
#pragma unroll
    for (int o = 16; o >= 1; o >>= 1) {
        s  += __shfl_xor(s,  o, 32);
        s2 += __shfl_xor(s2, o, 32);
    }
    int wid = tid >> 5, lane = tid & 31;
    if (lane == 0) { red[wid] = s; red[wid + 8] = s2; }
    __syncthreads();
    if (tid == 0) {
        float ts = 0.f, ts2 = 0.f;
        for (int w = 0; w < 8; ++w) { ts += red[w]; ts2 += red[w + 8]; }
        red[0] = ts; red[1] = ts2;
    }
    __syncthreads();
    float mean = red[0] * (1.0f / 512.0f);
    float var  = red[1] * (1.0f / 512.0f) - mean * mean;
    float rs   = rsqrtf(var + 1e-5f);
    float y0 = (x0 - mean) * rs * g[tid] + beta[tid];
    float y1 = (x1 - mean) * rs * g[tid + 256] + beta[tid + 256];
    outF[base + tid] = y0;
    outF[base + tid + 256] = y1;
    if (DUAL) {
        outB[base + tid] = f2bf(y0);
        outB[base + tid + 256] = f2bf(y1);
    }
}

// ---------------------------------------------------------------------------
// Host launch
// ---------------------------------------------------------------------------
extern "C" void kernel_launch(void* const* d_in, const int* in_sizes, int n_in,
                              void* d_out, int out_size, void* d_ws, size_t ws_size,
                              hipStream_t stream) {
    (void)in_sizes; (void)n_in; (void)out_size; (void)ws_size;
    const int S = 8192, D = 512, H = 2048;

    const float* enc  = (const float*)d_in[0];
    const float* prev = (const float*)d_in[1];
    const float* wq_m = (const float*)d_in[2];
    const float* wk_m = (const float*)d_in[3];
    const float* wv_m = (const float*)d_in[4];
    const float* wq_c = (const float*)d_in[5];
    const float* wk_c = (const float*)d_in[6];
    const float* wv_c = (const float*)d_in[7];
    const float* w_o  = (const float*)d_in[8];
    const float* ln_g = (const float*)d_in[9];
    const float* ln_b = (const float*)d_in[10];
    const float* w1   = (const float*)d_in[11];
    const float* b1   = (const float*)d_in[12];
    const float* w2   = (const float*)d_in[13];
    const float* b2   = (const float*)d_in[14];
    float* out = (float*)d_out;

    // ---- workspace bump allocator (byte offsets, 256B aligned) ----
    char* ws = (char*)d_ws;
    size_t cur = 0;
    auto take = [&](size_t bytes) { size_t o = cur; cur += (bytes + 255) & ~(size_t)255; return o; };
    bf16* prevB = (bf16*)(ws + take((size_t)S * D * 2));
    bf16* encB  = (bf16*)(ws + take((size_t)S * D * 2));
    bf16* WqkvM = (bf16*)(ws + take((size_t)D * 192 * 2));
    bf16* WkvC  = (bf16*)(ws + take((size_t)D * 128 * 2));
    bf16* WqC   = (bf16*)(ws + take((size_t)D * 64 * 2));
    bf16* Wsum  = (bf16*)(ws + take((size_t)64 * D * 2));
    bf16* W1B   = (bf16*)(ws + take((size_t)D * H * 2));
    bf16* W2B   = (bf16*)(ws + take((size_t)H * D * 2));
    bf16* QKVm  = (bf16*)(ws + take((size_t)S * 192 * 2));  // reused: KVc + Qc
    bf16* headM = (bf16*)(ws + take((size_t)S * 64 * 2));   // reused: headC
    bf16* mmhB  = (bf16*)(ws + take((size_t)S * D * 2));
    bf16* xB    = (bf16*)(ws + take((size_t)S * D * 2));
    bf16* hB    = (bf16*)(ws + take((size_t)S * H * 2));
    float* mhF  = (float*)(ws + take((size_t)S * D * 4));   // reused: ffn
    float* xF   = (float*)(ws + take((size_t)S * D * 4));
    bf16* KVc   = QKVm;                 // [S,128] (k|v), alive after QKVm is dead
    bf16* QcB   = QKVm + (size_t)S * 128;
    bf16* headC = headM;
    float* ffnF = mhF;

    // ---- 1) conversions / packing ----
    {
        size_t n = (size_t)S * D;
        f32_to_bf16_kernel<<<dim3((n + 255) / 256), 256, 0, stream>>>(prev, prevB, n);
        f32_to_bf16_kernel<<<dim3((n + 255) / 256), 256, 0, stream>>>(enc, encB, n);
    }
    f32_to_bf16_kernel<<<dim3(((size_t)D * 64 + 255) / 256), 256, 0, stream>>>(wq_c, WqC, (size_t)D * 64);
    f32_to_bf16_kernel<<<dim3(((size_t)D * H + 255) / 256), 256, 0, stream>>>(w1, W1B, (size_t)D * H);
    f32_to_bf16_kernel<<<dim3(((size_t)H * D + 255) / 256), 256, 0, stream>>>(w2, W2B, (size_t)H * D);
    pack_bf16_kernel<<<dim3((D * 192 + 255) / 256), 256, 0, stream>>>(wq_m, wk_m, wv_m, WqkvM, D, 64, 3);
    pack_bf16_kernel<<<dim3((D * 128 + 255) / 256), 256, 0, stream>>>(wk_c, wv_c, wv_c, WkvC, D, 64, 2);
    wsum_kernel<<<dim3((64 * 512 + 255) / 256), 256, 0, stream>>>(w_o, Wsum);

    // ---- 2) masked self-attention path ----
    gemm_bf16_kernel<1><<<dim3(S / 128, 192 / 64), 256, 0, stream>>>(
        prevB, WqkvM, nullptr, nullptr, QKVm, S, 192, D);
    attn_kernel<true><<<dim3(S / 128), 256, 0, stream>>>(
        QKVm, 192, QKVm + 64, QKVm + 128, 192, headM, S);
    gemm_bf16_kernel<1><<<dim3(S / 128, D / 64), 256, 0, stream>>>(
        headM, Wsum, nullptr, nullptr, mmhB, S, D, 64);

    // ---- 3) cross-attention path ----
    gemm_bf16_kernel<1><<<dim3(S / 128, 1), 256, 0, stream>>>(
        mmhB, WqC, nullptr, nullptr, QcB, S, 64, D);
    gemm_bf16_kernel<1><<<dim3(S / 128, 128 / 64), 256, 0, stream>>>(
        encB, WkvC, nullptr, nullptr, KVc, S, 128, D);
    attn_kernel<false><<<dim3(S / 128), 256, 0, stream>>>(
        QcB, 64, KVc, KVc + 64, 128, headC, S);
    gemm_bf16_kernel<0><<<dim3(S / 128, D / 64), 256, 0, stream>>>(
        headC, Wsum, nullptr, mhF, nullptr, S, D, 64);

    // ---- 4) x = LN(mh + prev) ----
    ln_residual_kernel<true><<<dim3(S), 256, 0, stream>>>(mhF, prev, ln_g, ln_b, xF, xB);

    // ---- 5) FFN ----
    gemm_bf16_kernel<2><<<dim3(S / 128, H / 64), 256, 0, stream>>>(
        xB, W1B, b1, nullptr, hB, S, H, D);
    gemm_bf16_kernel<3><<<dim3(S / 128, D / 64), 256, 0, stream>>>(
        hB, W2B, b2, ffnF, nullptr, S, D, H);

    // ---- 6) out = LN(ffn + x); outputs = (encoder_output, out) ----
    ln_residual_kernel<false><<<dim3(S), 256, 0, stream>>>(
        ffnF, xF, ln_g, ln_b, out + (size_t)S * D, nullptr);
    copy_f32_kernel<<<dim3(((size_t)S * D + 255) / 256), 256, 0, stream>>>(
        enc, out, (size_t)S * D);
}
